// TSpatialHyperGNN_12128987644191
// MI455X (gfx1250) — matrix operand order repaired and checked by reading, not compiled, not run on hardware
//
#include <hip/hip_runtime.h>
#include <hip/hip_bf16.h>
#include <math.h>

// TSpatialHyperGNN on MI455X (gfx1250), wave32 + WMMA bf16 path.
//   H1 = relu(X @ Wm + bm)                          (512x64)
//   Y  = einsum('ijk,jc,kc->ic', A, H, H)           (A: 512^3 fp32, streamed)
//   per i: M = A[i] @ H (WMMA bf16, fp32 acc), Y[i,c] = sum_j M[j,c]*H[j,c]
//   Z  = relu(Y @ W1 + b1) ; repeat conv ; head = Y2 @ W2 + b2 ; log_softmax

#define NN   512
#define DIN  128
#define DH   64
#define NCLS 40

#if defined(__AMDGCN__) && \
    __has_builtin(__builtin_amdgcn_global_load_async_to_lds_b128) && \
    __has_builtin(__builtin_amdgcn_s_wait_asynccnt)
#define USE_ASYNC_LDS 1
#else
#define USE_ASYNC_LDS 0
#endif

typedef __attribute__((ext_vector_type(16))) __bf16 v16bf;
typedef __attribute__((ext_vector_type(8)))  float  v8f;
typedef int async_v4i __attribute__((vector_size(16)));   // matches builtin pointee

// ---------------- MLP: H1 = relu(X @ Wm + bm), stored transposed bf16 -------
__global__ __launch_bounds__(DH)
void mlp_kernel(const float* __restrict__ X, const float* __restrict__ Wm,
                const float* __restrict__ bm, __bf16* __restrict__ HT) {
  __shared__ float sx[DIN];
  const int i = blockIdx.x, h = threadIdx.x;
  sx[h]      = X[i * DIN + h];
  sx[h + DH] = X[i * DIN + h + DH];
  __syncthreads();
  float acc = bm[h];
  #pragma unroll 8
  for (int d = 0; d < DIN; ++d) acc = fmaf(sx[d], Wm[d * DH + h], acc);
  acc = fmaxf(acc, 0.0f);
  HT[h * NN + i] = (__bf16)acc;   // HT[c][j], K-contiguous for WMMA B frags
}

// ---------------- heavy conv: Y[i,:] = sum_jk A[i,j,k] H[j,:] H[k,:] --------
__global__ __launch_bounds__(256)
void conv_kernel(const float* __restrict__ A, const __bf16* __restrict__ HTg,
                 float* __restrict__ Y) {
  __shared__ __bf16 sHT[DH * NN];              // 64 KB: H^T bf16, [c][k]

  const int i    = blockIdx.x;
  const int tid  = threadIdx.x;
  const int lane = tid & 31;
  const int wave = tid >> 5;                   // 0..7 (wave32)

  // stage H^T (64 KB) into LDS
#if USE_ASYNC_LDS
  {
    __attribute__((address_space(1))) async_v4i* gsrc =
        (__attribute__((address_space(1))) async_v4i*)HTg;
    __attribute__((address_space(3))) async_v4i* ldst =
        (__attribute__((address_space(3))) async_v4i*)sHT;
    #pragma unroll
    for (int v = 0; v < (DH * NN * 2 / 16) / 256; ++v) {
      const int idx = tid + v * 256;           // 16B elements
      __builtin_amdgcn_global_load_async_to_lds_b128(gsrc + idx, ldst + idx, 0, 0);
    }
    __builtin_amdgcn_s_wait_asynccnt(0);
  }
#else
  {
    const uint4* __restrict__ src = (const uint4*)HTg;
    uint4* dst = (uint4*)sHT;
    #pragma unroll
    for (int v = 0; v < (DH * NN * 2 / 16) / 256; ++v)
      dst[tid + v * 256] = src[tid + v * 256];
  }
#endif
  if (tid < DH) Y[(size_t)i * DH + tid] = 0.0f;  // zero my output row
  __syncthreads();

  const int cL     = lane & 15;                // col within 16-wide tile
  const int hiHalf = lane >> 4;                // 0: lanes 0-15, 1: lanes 16-31
  const int kbA    = hiHalf * 16;              // A-frag K sub-offset
  const int kbB    = hiHalf * 16;              // B-frag K sub-offset

  const float* __restrict__ Abase = A + (size_t)i * NN * NN;

  // each wave owns 4 consecutive j-tiles (wave*4 .. wave*4+3), all in regs
  const int jt0 = wave * 4;
  const float* Arow[4];
  #pragma unroll
  for (int u = 0; u < 4; ++u)
    Arow[u] = Abase + (size_t)((jt0 + u) * 16 + cL) * NN;

  v8f acc[4][4] = {};                          // [j-tile][c-tile] 16x16 fp32

  #pragma unroll 1
  for (int k0 = 0; k0 < NN; k0 += 32) {
    // B fragments for this k-chunk: 4 c-tiles, distinct registers, reused 4x
    v16bf b[4];
    #pragma unroll
    for (int ct = 0; ct < 4; ++ct)
      b[ct] = *(const v16bf*)&sHT[((ct << 4) + cL) * NN + k0 + kbB];

    // phase 1: load + convert ALL four A fragments (keeps the cvt writes a
    // full load-phase away from the WMMAs that read them -> no hazard nops)
    v16bf a[4];
    #pragma unroll
    for (int u = 0; u < 4; ++u) {
      __builtin_prefetch(Arow[u] + k0 + 128, 0, 3);
      const float4 f0 = *(const float4*)(Arow[u] + k0 + kbA);
      const float4 f1 = *(const float4*)(Arow[u] + k0 + kbA + 4);
      const float4 f2 = *(const float4*)(Arow[u] + k0 + kbA + 8);
      const float4 f3 = *(const float4*)(Arow[u] + k0 + kbA + 12);
      a[u][0]=(__bf16)f0.x;  a[u][1]=(__bf16)f0.y;  a[u][2]=(__bf16)f0.z;  a[u][3]=(__bf16)f0.w;
      a[u][4]=(__bf16)f1.x;  a[u][5]=(__bf16)f1.y;  a[u][6]=(__bf16)f1.z;  a[u][7]=(__bf16)f1.w;
      a[u][8]=(__bf16)f2.x;  a[u][9]=(__bf16)f2.y;  a[u][10]=(__bf16)f2.z; a[u][11]=(__bf16)f2.w;
      a[u][12]=(__bf16)f3.x; a[u][13]=(__bf16)f3.y; a[u][14]=(__bf16)f3.z; a[u][15]=(__bf16)f3.w;
    }

    // phase 2: 16-WMMA burst, 16 independent accumulator chains
    #pragma unroll
    for (int u = 0; u < 4; ++u)
      #pragma unroll
      for (int ct = 0; ct < 4; ++ct)
        acc[u][ct] = __builtin_amdgcn_wmma_f32_16x16x32_bf16(
            false, a[u], false, b[ct], (short)0, acc[u][ct], false, false);
  }

  // Y[i,c] += sum_j M[j,c] * H[j,c]  (C layout: VGPR r <-> row r (+8 hi half))
  #pragma unroll
  for (int u = 0; u < 4; ++u) {
    const int joff = (jt0 + u) * 16 + (hiHalf << 3);
    #pragma unroll
    for (int ct = 0; ct < 4; ++ct) {
      const __bf16* __restrict__ wp = &sHT[((ct << 4) + cL) * NN + joff];
      float p = 0.0f;
      #pragma unroll
      for (int r = 0; r < 8; ++r) p = fmaf(acc[u][ct][r], (float)wp[r], p);
      atomicAdd(&Y[(size_t)i * DH + (ct << 4) + cL], p);
    }
  }
}

// ---------------- Z = relu(Y @ W1 + b1), stored transposed bf16 -------------
__global__ __launch_bounds__(DH)
void lin_kernel(const float* __restrict__ Yin, const float* __restrict__ W,
                const float* __restrict__ b, __bf16* __restrict__ HT) {
  __shared__ float sy[DH];
  const int i = blockIdx.x, h = threadIdx.x;
  sy[h] = Yin[i * DH + h];
  __syncthreads();
  float acc = b[h];
  #pragma unroll 8
  for (int c = 0; c < DH; ++c) acc = fmaf(sy[c], W[c * DH + h], acc);
  acc = fmaxf(acc, 0.0f);
  HT[h * NN + i] = (__bf16)acc;
}

// ---------------- head: log_softmax(Y2 @ W2 + b2) ---------------------------
__global__ __launch_bounds__(DH)
void head_kernel(const float* __restrict__ Y2, const float* __restrict__ W2,
                 const float* __restrict__ b2, float* __restrict__ out) {
  __shared__ float sy[DH];
  __shared__ float slog[NCLS];
  __shared__ float sred[2];
  const int i = blockIdx.x, t = threadIdx.x;
  sy[t] = Y2[i * DH + t];
  __syncthreads();
  if (t < NCLS) {
    float acc = b2[t];
    #pragma unroll 8
    for (int c = 0; c < DH; ++c) acc = fmaf(sy[c], W2[c * NCLS + t], acc);
    slog[t] = acc;
  }
  __syncthreads();
  if (t == 0) {
    float m = slog[0];
    for (int n = 1; n < NCLS; ++n) m = fmaxf(m, slog[n]);
    float s = 0.0f;
    for (int n = 0; n < NCLS; ++n) s += expf(slog[n] - m);
    sred[0] = m; sred[1] = logf(s);
  }
  __syncthreads();
  if (t < NCLS) out[i * NCLS + t] = slog[t] - sred[0] - sred[1];
}

extern "C" void kernel_launch(void* const* d_in, const int* in_sizes, int n_in,
                              void* d_out, int out_size, void* d_ws, size_t ws_size,
                              hipStream_t stream) {
  const float* A  = (const float*)d_in[0];  // (512,512,512)
  const float* X  = (const float*)d_in[1];  // (512,128)
  const float* Wm = (const float*)d_in[2];  // (128,64)
  const float* bm = (const float*)d_in[3];  // (64,)
  const float* W1 = (const float*)d_in[4];  // (64,64)
  const float* b1 = (const float*)d_in[5];  // (64,)
  const float* W2 = (const float*)d_in[6];  // (64,40)
  const float* b2 = (const float*)d_in[7];  // (40,)
  float* out = (float*)d_out;               // (512,40)

  // workspace layout
  float*  Y1  = (float*)d_ws;                           // 512*64 f32
  float*  Y2  = Y1 + NN * DH;                           // 512*64 f32
  __bf16* HT1 = (__bf16*)((char*)d_ws + 2u * NN * DH * sizeof(float));
  __bf16* HT2 = HT1 + DH * NN;

  mlp_kernel <<<NN, DH,  0, stream>>>(X, Wm, bm, HT1);
  conv_kernel<<<NN, 256, 0, stream>>>(A, HT1, Y1);
  lin_kernel <<<NN, DH,  0, stream>>>(Y1, W1, b1, HT2);
  conv_kernel<<<NN, 256, 0, stream>>>(A, HT2, Y2);
  head_kernel<<<NN, DH,  0, stream>>>(Y2, W2, b2, out);
}
